// AttentionHead_67242007987106
// MI455X (gfx1250) — matrix operand order repaired and checked by reading, not compiled, not run on hardware
//
#include <hip/hip_runtime.h>
#include <math.h>

// Problem constants (from reference)
#define EMBED 1024
#define HD    64
#define BATCH 8
#define SEQ   2048
#define MT    (BATCH * SEQ)   // 16384 total rows
#define WPB   4               // waves per block (wave32)
#define NSPLIT 4              // key-dimension split for the attention pass
#define KSPAN (SEQ / NSPLIT)  // 512 keys per partial

typedef __attribute__((ext_vector_type(16))) _Float16 v16h;
typedef __attribute__((ext_vector_type(8)))  float    v8f;

static __device__ __forceinline__ v8f wmma_f16(v16h a, v16h b, v8f c) {
  // D = A(16x32 f16) x B(32x16 f16) + C(16x16 f32)
  return __builtin_amdgcn_wmma_f32_16x16x32_f16(false, a, false, b, (short)0, c,
                                                false, false);
}

// ---------- tile loaders (layouts per CDNA5 ISA 7.12.2) ----------
// A-matrix 16x32 f16: lane L holds row M=L%16; half h -> K = h + 16*(L>=16) (this code's order, same as B)
// => per lane: 16 contiguous elements [base..base+15].
static __device__ __forceinline__ v16h pack_a_f32(const float* __restrict__ p) {
  v16h a;
#pragma unroll
  for (int i = 0; i < 8; ++i) {
    a[i]     = (_Float16)p[i];
    a[i + 8] = (_Float16)p[8 + i];
  }
  return a;
}
// B-matrix 32x16 f16: lane L holds col N=L%16; half h -> K = h + 16*(L>=16) (this code's order, not the ISA table's)
// => per lane: 16 contiguous elements.
static __device__ __forceinline__ v16h pack_b_f32(const float* __restrict__ p) {
  v16h b;
#pragma unroll
  for (int i = 0; i < 16; ++i) b[i] = (_Float16)p[i];
  return b;
}
static __device__ __forceinline__ v16h load_b_f16(const _Float16* __restrict__ p) {
  return *(const v16h*)p;  // 32B contiguous -> 2x global_load_b128
}
static __device__ __forceinline__ v16h load_a_f16(const _Float16* __restrict__ p) {
  v16h a;
#pragma unroll
  for (int i = 0; i < 8; ++i) {
    a[i]     = p[i];
    a[i + 8] = p[8 + i];
  }
  return a;
}

// =====================================================================
// Kernel 1: one projection GEMM per wave (p = 0:Q, 1:K, 2:V).
// 3072 waves total; no cross-projection reuse exists, so splitting is free
// and cuts register pressure (4 accumulators instead of 12).
//   qh[m][d] = ((query @ Wq^T)[m][d] + bq[d]) * rsqrt(HD)   (scale folded in)
//   kh[m][d] =  (key   @ Wk^T)[m][d] + bk[d]
//   vt[d][m] =  (value @ Wv^T)[m][d] + bv[d]                (V stored transposed)
// =====================================================================
__global__ void __launch_bounds__(WPB * 32)
qkv_proj(const float* __restrict__ Xq, const float* __restrict__ Xk,
         const float* __restrict__ Xv,
         const float* __restrict__ Wq, const float* __restrict__ bq,
         const float* __restrict__ Wk, const float* __restrict__ bk,
         const float* __restrict__ Wv, const float* __restrict__ bv,
         _Float16* __restrict__ qh, _Float16* __restrict__ kh,
         _Float16* __restrict__ vt) {
  const int lane = threadIdx.x & 31;
  const int wave = threadIdx.x >> 5;
  const int wid  = blockIdx.x * WPB + wave;   // 0 .. 3*MT/16-1
  if (wid >= 3 * (MT / 16)) return;
  const int p    = wid >> 10;                 // MT/16 == 1024
  const int tile = wid & 1023;
  const int m0   = tile * 16;
  const int half = lane >> 4;
  const int ln   = lane & 15;

  const float* X = (p == 0) ? Xq : (p == 1) ? Xk : Xv;
  const float* W = (p == 0) ? Wq : (p == 1) ? Wk : Wv;
  const float* bias = (p == 0) ? bq : (p == 1) ? bk : bv;

  v8f acc[4] = {{}, {}, {}, {}};

  const size_t arow = (size_t)(m0 + ln) * EMBED;  // A row for this lane
  const int aoff = half * 16;                     // A half-offset in K
  const int boff = half * 16;                     // B half-offset in K

  for (int e0 = 0; e0 < EMBED; e0 += 32) {
    const v16h A = pack_a_f32(X + arow + e0 + aoff);
#pragma unroll
    for (int nt = 0; nt < 4; ++nt) {
      const size_t wrow = (size_t)(nt * 16 + ln) * EMBED + e0 + boff;
      acc[nt] = wmma_f16(A, pack_b_f32(W + wrow), acc[nt]);
    }
  }

  // Epilogue: C/D layout -> lane col N = ln, VGPR r -> row M = r + 8*half
#pragma unroll
  for (int nt = 0; nt < 4; ++nt) {
    const int col = nt * 16 + ln;
    const float bv_ = bias[col];
#pragma unroll
    for (int r = 0; r < 8; ++r) {
      const int m = m0 + r + half * 8;
      const float v = acc[nt][r] + bv_;
      if (p == 0)      qh[(size_t)m * HD + col] = (_Float16)(v * 0.125f);
      else if (p == 1) kh[(size_t)m * HD + col] = (_Float16)v;
      else             vt[(size_t)col * MT + m] = (_Float16)v;
    }
  }
}

// =====================================================================
// Kernel 2: flash attention, key-split. One wave per (part, batch, 16-query
// tile) => 4096 waves. Each wave streams 512 keys in chunks of 32 with online
// softmax and writes unnormalized partial O plus per-row (m, l).
// =====================================================================
__global__ void __launch_bounds__(WPB * 32)
attn_partial(const _Float16* __restrict__ qh, const _Float16* __restrict__ kh,
             const _Float16* __restrict__ vt, const int* __restrict__ kmask,
             float* __restrict__ op, float* __restrict__ mp,
             float* __restrict__ lp) {
  __shared__ _Float16 lds[WPB][16 * 32];  // per-wave 16x32 P staging tile

  const int lane = threadIdx.x & 31;
  const int wave = threadIdx.x >> 5;
  const int tid  = blockIdx.x * WPB + wave;  // 0 .. NSPLIT*B*(S/16)-1
  if (tid >= NSPLIT * BATCH * (SEQ / 16)) return;
  const int part = tid >> 10;        // B*(S/16) == 1024
  const int qg   = tid & 1023;
  const int b    = qg >> 7;          // SEQ/16 == 128
  const int m0   = (qg & 127) * 16;
  const int half = lane >> 4;
  const int ln   = lane & 15;
  const float NEGINF = -__builtin_inff();

  // Q A-tiles for head-dim K-steps d0 = 0, 32 — resident all pass.
  const _Float16* qrow = qh + (size_t)(b * SEQ + m0 + ln) * HD + half * 16;
  const v16h A0 = load_a_f16(qrow);
  const v16h A1 = load_a_f16(qrow + 32);

  float mrow[8], lrow[8];
  v8f o[4] = {{}, {}, {}, {}};
#pragma unroll
  for (int r = 0; r < 8; ++r) { mrow[r] = NEGINF; lrow[r] = 0.f; }

  _Float16* buf = lds[wave];
  const int kbeg = part * KSPAN;

  for (int c0 = kbeg; c0 < kbeg + KSPAN; c0 += 32) {
    // ---- scores: two 16(query) x 16(key) tiles ----
    v8f s[2];
#pragma unroll
    for (int sub = 0; sub < 2; ++sub) {
      const int kb = c0 + sub * 16;
      // B[d][n] = kh[n][d]; lane col n = kb+ln, K = d = h + 16*half (+k-step)
      const _Float16* krow = kh + (size_t)(b * SEQ + kb + ln) * HD + half * 16;
      v8f acc = {};
      acc = wmma_f16(A0, load_b_f16(krow), acc);
      acc = wmma_f16(A1, load_b_f16(krow + 32), acc);
      const float biasm = kmask[b * SEQ + kb + ln] ? 0.f : NEGINF;
#pragma unroll
      for (int r = 0; r < 8; ++r) acc[r] += biasm;
      s[sub] = acc;
    }

    // ---- online softmax: rows live in 16-lane halves of the D layout ----
    float mx[8];
#pragma unroll
    for (int r = 0; r < 8; ++r) mx[r] = fmaxf(s[0][r], s[1][r]);
#pragma unroll
    for (int off = 1; off <= 8; off <<= 1)
#pragma unroll
      for (int r = 0; r < 8; ++r) mx[r] = fmaxf(mx[r], __shfl_xor(mx[r], off));

    float muse[8], scl[8];
#pragma unroll
    for (int r = 0; r < 8; ++r) {
      const float mnew = fmaxf(mrow[r], mx[r]);
      muse[r] = (mnew == NEGINF) ? 0.f : mnew;  // fully-masked-so-far guard
      scl[r]  = __expf(mrow[r] - muse[r]);
      mrow[r] = mnew;
    }
#pragma unroll
    for (int sub = 0; sub < 2; ++sub)
#pragma unroll
      for (int r = 0; r < 8; ++r) s[sub][r] = __expf(s[sub][r] - muse[r]);

    float rs[8];
#pragma unroll
    for (int r = 0; r < 8; ++r) rs[r] = s[0][r] + s[1][r];
#pragma unroll
    for (int off = 1; off <= 8; off <<= 1)
#pragma unroll
      for (int r = 0; r < 8; ++r) rs[r] += __shfl_xor(rs[r], off);
#pragma unroll
    for (int r = 0; r < 8; ++r) lrow[r] = lrow[r] * scl[r] + rs[r];
#pragma unroll
    for (int nt = 0; nt < 4; ++nt)
#pragma unroll
      for (int r = 0; r < 8; ++r) o[nt][r] *= scl[r];

    // ---- transpose P: D layout (row r+8*half, col ln) -> A layout via LDS ----
#pragma unroll
    for (int sub = 0; sub < 2; ++sub)
#pragma unroll
      for (int r = 0; r < 8; ++r)
        buf[(r + half * 8) * 32 + sub * 16 + ln] = (_Float16)s[sub][r];
    asm volatile("s_wait_dscnt 0" ::: "memory");  // order ds stores before reads
    const v16h pA = load_a_f16(buf + ln * 32 + half * 16);

    // ---- o += P @ V : B[kk][n] = vt[nt*16+n][c0+kk], contiguous in kk ----
#pragma unroll
    for (int nt = 0; nt < 4; ++nt) {
      const _Float16* vrow =
          vt + (size_t)(nt * 16 + ln) * MT + b * SEQ + c0 + half * 16;
      o[nt] = wmma_f16(pA, load_b_f16(vrow), o[nt]);
    }
  }

  // ---- write partials: unnormalized O, per-row (m, l) ----
#pragma unroll
  for (int nt = 0; nt < 4; ++nt) {
#pragma unroll
    for (int r = 0; r < 8; ++r) {
      const int g = b * SEQ + m0 + r + half * 8;  // global row
      op[(size_t)part * MT * HD + (size_t)g * HD + nt * 16 + ln] = o[nt][r];
    }
  }
  if (ln == 0) {  // stats are uniform across each 16-lane half
#pragma unroll
    for (int r = 0; r < 8; ++r) {
      const int g = b * SEQ + m0 + r + half * 8;
      mp[part * MT + g] = mrow[r];
      lp[part * MT + g] = lrow[r];
    }
  }
}

// =====================================================================
// Kernel 3: combine the NSPLIT partials (log-sum-exp merge) and normalize.
// One thread per output element.
// =====================================================================
__global__ void __launch_bounds__(256)
attn_reduce(const float* __restrict__ op, const float* __restrict__ mp,
            const float* __restrict__ lp, const int* __restrict__ qmask,
            float* __restrict__ out) {
  const int idx = blockIdx.x * 256 + threadIdx.x;
  if (idx >= MT * HD) return;
  const int g = idx / HD;  // global row (b*SEQ + s)
  const float NEGINF = -__builtin_inff();

  float M = NEGINF;
#pragma unroll
  for (int p_ = 0; p_ < NSPLIT; ++p_) M = fmaxf(M, mp[p_ * MT + g]);
  const float Mu = (M == NEGINF) ? 0.f : M;

  float L = 0.f, O = 0.f;
#pragma unroll
  for (int p_ = 0; p_ < NSPLIT; ++p_) {
    const float w = __expf(mp[p_ * MT + g] - Mu);
    L += lp[p_ * MT + g] * w;
    O += op[(size_t)p_ * MT * HD + idx] * w;
  }
  // query_mask==0 rows -> NaN (matches reference softmax of all -inf)
  out[idx] = qmask[g] ? (O / L) : __builtin_nanf("");
}

// =====================================================================
extern "C" void kernel_launch(void* const* d_in, const int* in_sizes, int n_in,
                              void* d_out, int out_size, void* d_ws, size_t ws_size,
                              hipStream_t stream) {
  (void)in_sizes; (void)n_in; (void)out_size; (void)ws_size;
  const float* query = (const float*)d_in[0];
  const float* key   = (const float*)d_in[1];
  const float* value = (const float*)d_in[2];
  const int* qmask   = (const int*)d_in[3];
  const int* kmask   = (const int*)d_in[4];
  const float* Wq = (const float*)d_in[5];
  const float* bq = (const float*)d_in[6];
  const float* Wk = (const float*)d_in[7];
  const float* bk = (const float*)d_in[8];
  const float* Wv = (const float*)d_in[9];
  const float* bv = (const float*)d_in[10];
  float* out = (float*)d_out;

  // Workspace layout (~22.5 MB):
  //   qh,kh (row-major f16) + vt (transposed f16): 3 * 2 MB
  //   op: NSPLIT*MT*HD f32 = 16 MB ; mp, lp: NSPLIT*MT f32 = 256 KB each
  _Float16* qh = (_Float16*)d_ws;
  _Float16* kh = qh + (size_t)MT * HD;
  _Float16* vt = kh + (size_t)MT * HD;
  float* op = (float*)(vt + (size_t)MT * HD);
  float* mp = op + (size_t)NSPLIT * MT * HD;
  float* lp = mp + (size_t)NSPLIT * MT;

  dim3 blk(WPB * 32);

  const int ptiles = 3 * (MT / 16);  // 3072 wave-tiles (proj x row-tile)
  qkv_proj<<<dim3((ptiles + WPB - 1) / WPB), blk, 0, stream>>>(
      query, key, value, Wq, bq, Wk, bk, Wv, bv, qh, kh, vt);

  const int atiles = NSPLIT * BATCH * (SEQ / 16);  // 4096 wave-tiles
  attn_partial<<<dim3((atiles + WPB - 1) / WPB), blk, 0, stream>>>(
      qh, kh, vt, kmask, op, mp, lp);

  const int n = MT * HD;  // 1M output elements
  attn_reduce<<<dim3((n + 255) / 256), dim3(256), 0, stream>>>(op, mp, lp,
                                                               qmask, out);
}
